// Quantize_37512244363876
// MI455X (gfx1250) — matrix operand order repaired and checked by reading, not compiled, not run on hardware
//
#include <hip/hip_runtime.h>
#include <hip/hip_bf16.h>
#include <cstdint>

#define DIM   256
#define NEMB  512
#define NROWS (32 * 4096)          // 131072 flattened input rows
#define ROWS_PER_BLOCK 128
#define CHUNK 64                   // codebook entries per TDM chunk (double-buffered)
#define NCHUNK (NEMB / CHUNK)      // 8
#define LSTRIDE 264                // bf16 elems per LDS row (256 + 8 pad -> 528B, conflict-free)

typedef __attribute__((ext_vector_type(16))) __bf16 v16bf;
typedef __attribute__((ext_vector_type(8)))  float  v8f;
typedef __attribute__((ext_vector_type(4)))  unsigned int u32x4;
typedef __attribute__((ext_vector_type(8)))  int i32x8;
typedef __attribute__((ext_vector_type(4)))  int i32x4;

union BV { v16bf v; uint4 q[2]; };

// ---------------------------------------------------------------------------
// Kernel 1: normalize codebook rows; write fp32 embed_n + bf16 hi/lo split.
// ---------------------------------------------------------------------------
__global__ __launch_bounds__(256)
void vq_prep_kernel(const float* __restrict__ emb,
                    float* __restrict__ embed_n,
                    __bf16* __restrict__ cb_hi,
                    __bf16* __restrict__ cb_lo)
{
    __shared__ float red[256];
    const int e = blockIdx.x, t = threadIdx.x;
    float v = emb[(size_t)e * DIM + t];
    red[t] = v * v;
    __syncthreads();
    for (int s = 128; s > 0; s >>= 1) {
        if (t < s) red[t] += red[t + s];
        __syncthreads();
    }
    float en = v / sqrtf(red[0]);
    embed_n[(size_t)e * DIM + t] = en;
    __bf16 hh = (__bf16)en;
    cb_hi[(size_t)e * DIM + t] = hh;
    cb_lo[(size_t)e * DIM + t] = (__bf16)(en - (float)hh);
}

// ---------------------------------------------------------------------------
// Kernel 2: gram = || sum_e embed_n[e] ||^2 (entropy collapses to this);
// also zero the deterministic fixed-point diff accumulator.
// ---------------------------------------------------------------------------
__global__ __launch_bounds__(256)
void vq_gram_kernel(const float* __restrict__ embed_n,
                    float* __restrict__ gram,
                    unsigned long long* __restrict__ diff_acc)
{
    __shared__ float red[256];
    const int t = threadIdx.x;
    float s = 0.0f;
    for (int e = 0; e < NEMB; ++e) s += embed_n[(size_t)e * DIM + t];
    red[t] = s * s;
    __syncthreads();
    for (int k = 128; k > 0; k >>= 1) {
        if (t < k) red[t] += red[t + k];
        __syncthreads();
    }
    if (t == 0) { *gram = red[0]; *diff_acc = 0ULL; }
}

// LDS byte offset of a __shared__ object (flat addr low 32 bits = LDS address)
__device__ __forceinline__ unsigned lds_off(const void* p) {
    return (unsigned)(uintptr_t)p;
}

// Issue one TDM load: 8 rows x 256 bf16 tile, global row-major -> LDS with
// 16B padding after every 512B row (LSTRIDE layout).  Per-wave uniform args.
__device__ __forceinline__ void tdm_issue(unsigned ldsaddr, const __bf16* gsrc) {
    unsigned long long ga = (unsigned long long)(uintptr_t)gsrc;
    u32x4 g0;
    g0[0] = 1u;                                        // count=1 (valid user D#)
    g0[1] = ldsaddr;                                   // lds_addr (bytes)
    g0[2] = (unsigned)(ga & 0xFFFFFFFFu);              // global_addr[31:0]
    g0[3] = (unsigned)((ga >> 32) & 0x01FFFFFFu)       // global_addr[56:32]
          | (2u << 30);                                // type = 2 (image)
    i32x8 g1;
    g1[0] = (1 << 16)        // data_size = 2 bytes
          | (1 << 20)        // pad_enable
          | (6 << 22)        // pad_interval: every 512B (one row)
          | (3 << 25);       // pad_amount: 4 dwords = 16B
    g1[1] = (int)(256u << 16);   // tensor_dim0 = 256 (low 16 -> bits[31:16])
    g1[2] = (int)(8u << 16);     // tensor_dim1 = 8 rows (low 16 -> bits[31:16])
    g1[3] = (int)(256u << 16);   // tile_dim0 = 256 (bits[127:112])
    g1[4] = 8;                   // tile_dim1 = 8 (bits[143:128]); tile_dim2 = 0
    g1[5] = 256;                 // tensor_dim0_stride = 256 elems
    g1[6] = 0;
    g1[7] = 0;
    i32x4 z4 = {0, 0, 0, 0};             // groups 2/3 unused (2-D tile)
    i32x8 z8 = {0, 0, 0, 0, 0, 0, 0, 0}; // extra group (6-arg toolchain form)
    __builtin_amdgcn_tensor_load_to_lds(g0, g1, z4, z4, z8, 0);
}

// ---------------------------------------------------------------------------
// Kernel 3: main — WMMA bf16 (hi/lo split ~ fp32 accuracy) scores + argmax +
// gather + out write + diff accumulation.  Codebook streamed into LDS by the
// Tensor Data Mover, double-buffered so DMA overlaps WMMA compute.
//   block = 256 threads = 8 waves; each wave owns a 16-row M-tile and DMAs an
//   8-row slice of each codebook chunk (own TENSORcnt per wave).
// ---------------------------------------------------------------------------
__global__ __launch_bounds__(256)
void vq_main_kernel(const float* __restrict__ X,
                    const float* __restrict__ embed_n,
                    const __bf16* __restrict__ cb_hi,
                    const __bf16* __restrict__ cb_lo,
                    float* __restrict__ out,
                    float* __restrict__ ind_out,
                    unsigned long long* __restrict__ diff_acc)
{
    __shared__ __align__(16) __bf16 s_cb[2][2][CHUNK * LSTRIDE];  // [buf][hi/lo] 132KB
    __shared__ float s_red[8];

    const int tid  = threadIdx.x;
    const int lane = tid & 31;
    const int wave = tid >> 5;
    const int h    = lane >> 4;   // K-half per WMMA A/B 16-bit layout
    const int m    = lane & 15;
    const size_t rowBase = (size_t)blockIdx.x * ROWS_PER_BLOCK + (size_t)wave * 16;

    // prologue: DMA chunk 0 into buffer 0 (overlaps with A-fragment build below)
    const unsigned sliceOff = (unsigned)(wave * 8 * LSTRIDE);
    tdm_issue(lds_off(&s_cb[0][0][sliceOff]), cb_hi + (size_t)wave * 8 * DIM);
    tdm_issue(lds_off(&s_cb[0][1][sliceOff]), cb_lo + (size_t)wave * 8 * DIM);

    // ---- build A fragments (16 rows x K=256) with hi/lo bf16 split ----
    // 16-bit A layout: lanes 0-15 hold K {c..c+7, c+16..c+23}; lanes 16-31 the rest.
    const float* xrow = X + (rowBase + m) * DIM;
    v16bf a_hi[8], a_lo[8];
#pragma unroll
    for (int kc = 0; kc < 8; ++kc) {
        const int b1 = kc * 32 + 8 * h;
        const int b2 = b1 + 16;
        float4 f0 = *(const float4*)(xrow + b1);
        float4 f1 = *(const float4*)(xrow + b1 + 4);
        float4 f2 = *(const float4*)(xrow + b2);
        float4 f3 = *(const float4*)(xrow + b2 + 4);
        float fs[16] = {f0.x,f0.y,f0.z,f0.w, f1.x,f1.y,f1.z,f1.w,
                        f2.x,f2.y,f2.z,f2.w, f3.x,f3.y,f3.z,f3.w};
#pragma unroll
        for (int i = 0; i < 16; ++i) {
            __bf16 hh = (__bf16)fs[i];
            a_hi[kc][i] = hh;
            a_lo[kc][i] = (__bf16)(fs[i] - (float)hh);
        }
    }

    float best[8];
    int   bidx[8];
#pragma unroll
    for (int j = 0; j < 8; ++j) { best[j] = -3.4e38f; bidx[j] = 0; }

    __builtin_amdgcn_s_wait_tensorcnt(0);   // chunk 0 resident
    __syncthreads();

#pragma unroll 1
    for (int ch = 0; ch < NCHUNK; ++ch) {
        const int cur = ch & 1;
        const int nb  = cur ^ 1;
        if (ch + 1 < NCHUNK) {   // overlap next chunk's DMA with this chunk's WMMAs
            const size_t gofs = ((size_t)(ch + 1) * CHUNK + (size_t)wave * 8) * DIM;
            tdm_issue(lds_off(&s_cb[nb][0][sliceOff]), cb_hi + gofs);
            tdm_issue(lds_off(&s_cb[nb][1][sliceOff]), cb_lo + gofs);
        }

#pragma unroll 1
        for (int nt = 0; nt < CHUNK / 16; ++nt) {
            v8f c = {};
            // B layout: lane n holds column (nt*16 + n%16), K = {8*h..8*h+7, 16+8*h..16+8*h+7}
            const __bf16* bcol_h = &s_cb[cur][0][(size_t)(nt * 16 + m) * LSTRIDE + 8 * h];
            const __bf16* bcol_l = &s_cb[cur][1][(size_t)(nt * 16 + m) * LSTRIDE + 8 * h];
#pragma unroll
            for (int kc = 0; kc < 8; ++kc) {
                BV bh, bl;
                const __bf16* ph = bcol_h + kc * 32;
                const __bf16* pl = bcol_l + kc * 32;
                bh.q[0] = *(const uint4*)(ph);
                bh.q[1] = *(const uint4*)(ph + 16);
                bl.q[0] = *(const uint4*)(pl);
                bl.q[1] = *(const uint4*)(pl + 16);
                // score = x_hi*e_hi + x_lo*e_hi + x_hi*e_lo   (~fp32-accurate)
                c = __builtin_amdgcn_wmma_f32_16x16x32_bf16(false, a_hi[kc], false, bh.v, (short)0, c, false, false);
                c = __builtin_amdgcn_wmma_f32_16x16x32_bf16(false, a_lo[kc], false, bh.v, (short)0, c, false, false);
                c = __builtin_amdgcn_wmma_f32_16x16x32_bf16(false, a_hi[kc], false, bl.v, (short)0, c, false, false);
            }
            const int colBase = ch * CHUNK + nt * 16 + m;
#pragma unroll
            for (int j = 0; j < 8; ++j) {     // C vgpr j: M=j (lanes 0-15), M=j+8 (lanes 16-31)
                float cv = c[j];
                if (cv > best[j]) { best[j] = cv; bidx[j] = colBase; }
            }
        }

        __builtin_amdgcn_s_wait_tensorcnt(0);   // this wave's next-chunk DMA done
        __syncthreads();                        // everyone's slices visible
    }

    // cross-lane argmax within each 16-lane half (prefer lower index on ties)
#pragma unroll
    for (int j = 0; j < 8; ++j) {
        float v = best[j]; int i = bidx[j];
#pragma unroll
        for (int off = 1; off < 16; off <<= 1) {
            float ov = __shfl_xor(v, off, 32);
            int   oi = __shfl_xor(i, off, 32);
            if (ov > v || (ov == v && oi < i)) { v = ov; i = oi; }
        }
        bidx[j] = i;
    }

    // gather normalized code rows, write out + indices, accumulate diff
    float dacc = 0.0f;
#pragma unroll 1
    for (int r = 0; r < 16; ++r) {
        const int j = r & 7;
        const int src = (r < 8) ? 0 : 16;
        const int idx = __shfl(bidx[j], src, 32);
        const float* qrow = embed_n + (size_t)idx * DIM;
        const float* xr   = X   + (rowBase + r) * DIM;
        float*       orow = out + (rowBase + r) * DIM;
#pragma unroll
        for (int t = 0; t < DIM / 32; ++t) {
            int cidx = lane + t * 32;
            float qv = qrow[cidx];
            float xv = xr[cidx];
            orow[cidx] = qv;
            float d = qv - xv;
            dacc += d * d;
        }
        if (lane == 0) ind_out[rowBase + r] = (float)idx;
    }

    // deterministic diff reduction: wave shuffle -> LDS -> u64 fixed-point atomic
#pragma unroll
    for (int off = 16; off >= 1; off >>= 1) dacc += __shfl_xor(dacc, off, 32);
    if (lane == 0) s_red[wave] = dacc;
    __syncthreads();
    if (tid == 0) {
        float s = 0.0f;
        for (int w = 0; w < 8; ++w) s += s_red[w];
        atomicAdd(diff_acc, (unsigned long long)((double)s * 1048576.0));
    }
}

// ---------------------------------------------------------------------------
// Kernel 4: loss = diff - (1/E)^2 * entropy,  entropy = 2E^2 - 2*gram
// ---------------------------------------------------------------------------
__global__ void vq_loss_kernel(const unsigned long long* __restrict__ diff_acc,
                               const float* __restrict__ gram,
                               float* __restrict__ loss_out)
{
    double diff = (double)(*diff_acc) * (1.0 / 1048576.0) / ((double)NROWS * (double)DIM);
    double g = (double)(*gram);
    double E = (double)NEMB;
    double entropy = 2.0 * E * E - 2.0 * g;
    loss_out[0] = (float)(diff - entropy / (E * E));
}

// ---------------------------------------------------------------------------
extern "C" void kernel_launch(void* const* d_in, const int* in_sizes, int n_in,
                              void* d_out, int out_size, void* d_ws, size_t ws_size,
                              hipStream_t stream) {
    (void)in_sizes; (void)n_in; (void)out_size; (void)ws_size;
    const float* X   = (const float*)d_in[0];   // [32,4096,256] fp32
    const float* emb = (const float*)d_in[1];   // [512,256] fp32

    char* ws = (char*)d_ws;
    float*  embed_n = (float*)ws;                         // 512*256*4 = 524288 B
    __bf16* cb_hi   = (__bf16*)(ws + 524288);             // 262144 B
    __bf16* cb_lo   = (__bf16*)(ws + 786432);             // 262144 B
    float*  gram    = (float*)(ws + 1048576);
    unsigned long long* diffa = (unsigned long long*)(ws + 1048584);

    float* outp     = (float*)d_out;
    float* out_mat  = outp;                               // [N, D]
    float* out_loss = outp + (size_t)NROWS * DIM;         // scalar
    float* out_ind  = out_loss + 1;                       // [N] indices (as float)

    vq_prep_kernel<<<NEMB, 256, 0, stream>>>(emb, embed_n, cb_hi, cb_lo);
    vq_gram_kernel<<<1, 256, 0, stream>>>(embed_n, gram, diffa);
    vq_main_kernel<<<NROWS / ROWS_PER_BLOCK, 256, 0, stream>>>(
        X, embed_n, cb_hi, cb_lo, out_mat, out_ind, diffa);
    vq_loss_kernel<<<1, 1, 0, stream>>>(diffa, gram, out_loss);
}
